// TriMemoryBlock_11596411699812
// MI455X (gfx1250) — matrix operand order repaired and checked by reading, not compiled, not run on hardware
//
#include <hip/hip_runtime.h>
#include <hip/hip_bf16.h>

// ---------------------------------------------------------------------------
// TriMemoryBlock on MI455X (gfx1250): bf16 WMMA GEMMs with async-copy
// (GLOBAL_LOAD_ASYNC_TO_LDS) double-buffered weight tiles + windowed
// attention with in-place 16-bit LDS softmax.
// ---------------------------------------------------------------------------

#define BB    2
#define TT    2048
#define DD    1024
#define DFF_  4096
#define NH    16
#define HD    64
#define WW    512
#define MTOT  (BB * TT)          // 4096 activation rows
#define KBMAX 33                 // max key blocks per 512-window (+1 partial)
#define SCB   546                // 16-bit score/prob row stride (273 dwords, odd)
#define WPAD  40                 // LDS W-tile row stride (elements)

typedef __attribute__((ext_vector_type(16))) __bf16 v16bf;
typedef __attribute__((ext_vector_type(8)))  float  v8f;

__device__ __forceinline__ __bf16 f2bf(float f) {
  unsigned u = __builtin_bit_cast(unsigned, f);
  u += 0x7FFFu + ((u >> 16) & 1u);              // round-to-nearest-even
  unsigned short h = (unsigned short)(u >> 16);
  return __builtin_bit_cast(__bf16, h);
}
__device__ __forceinline__ unsigned short bfbits(float f) {
  return __builtin_bit_cast(unsigned short, f2bf(f));
}
__device__ __forceinline__ float bf2f(__bf16 b) {
  unsigned u = ((unsigned)__builtin_bit_cast(unsigned short, b)) << 16;
  return __builtin_bit_cast(float, u);
}
__device__ __forceinline__ unsigned short f2h_bits(float f) {
  _Float16 h = (_Float16)f;
  return __builtin_bit_cast(unsigned short, h);
}
__device__ __forceinline__ float h2f(unsigned short u) {
  return (float)__builtin_bit_cast(_Float16, u);
}
__device__ __forceinline__ v8f vzero8() {
  v8f z;
#pragma unroll
  for (int i = 0; i < 8; ++i) z[i] = 0.f;
  return z;
}

// Async global->LDS copy of 16 bytes per lane (CDNA5 GLOBAL_LOAD_ASYNC_TO_LDS,
// tracked by ASYNCcnt). ldsOff = LDS byte offset (low 32 bits of the generic
// LDS pointer), g = 64-bit global address.
__device__ __forceinline__ void async_ld16(unsigned ldsOff, const __bf16* g) {
  asm volatile("global_load_async_to_lds_b128 %0, %1, off"
               :: "v"(ldsOff), "v"(g) : "memory");
}
__device__ __forceinline__ void wait_async0() {
  asm volatile("s_wait_asynccnt 0x0" ::: "memory");
}

// ---------------------------------------------------------------------------
// f32 -> bf16, 4 elements/thread (float4 in, packed uint2 out)
// ---------------------------------------------------------------------------
__global__ __launch_bounds__(256) void f32_to_bf16_kernel(const float4* __restrict__ in,
                                                          uint2* __restrict__ out, int n4) {
  int i = blockIdx.x * 256 + threadIdx.x;
  if (i < n4) {
    float4 f = in[i];
    uint2 o;
    o.x = (unsigned)bfbits(f.x) | ((unsigned)bfbits(f.y) << 16);
    o.y = (unsigned)bfbits(f.z) | ((unsigned)bfbits(f.w) << 16);
    out[i] = o;
  }
}

// ---------------------------------------------------------------------------
// RMSNorm: one block per row of D=1024, writes f32 (optional) + bf16.
// ---------------------------------------------------------------------------
template <bool WRITE_F32>
__global__ __launch_bounds__(256) void rmsnorm_kernel(const float* __restrict__ x,
                                                      const float* __restrict__ w,
                                                      float* __restrict__ of,
                                                      __bf16* __restrict__ ob) {
  __shared__ float red[8];
  const int row = blockIdx.x;
  const float* xr = x + (size_t)row * DD;
  float vals[4];
  float s = 0.f;
#pragma unroll
  for (int i = 0; i < 4; ++i) {
    vals[i] = xr[threadIdx.x + i * 256];
    s += vals[i] * vals[i];
  }
#pragma unroll
  for (int off = 16; off > 0; off >>= 1) s += __shfl_xor(s, off);
  const int lane = threadIdx.x & 31, wv = threadIdx.x >> 5;
  if (lane == 0) red[wv] = s;
  __syncthreads();
  if (wv == 0) {
    float t = (lane < 8) ? red[lane] : 0.f;
#pragma unroll
    for (int off = 4; off > 0; off >>= 1) t += __shfl_xor(t, off);
    if (lane == 0) red[0] = t;
  }
  __syncthreads();
  const float inv = rsqrtf(red[0] * (1.f / DD) + 1e-6f);
#pragma unroll
  for (int i = 0; i < 4; ++i) {
    const int c = threadIdx.x + i * 256;
    const float o = vals[i] * inv * w[c];
    if (WRITE_F32) of[(size_t)row * DD + c] = o;
    ob[(size_t)row * DD + c] = f2bf(o);
  }
}

// ---------------------------------------------------------------------------
// bf16 WMMA GEMM: C[M,N] = A[M,K] @ W[N,K]^T (+epilogue).
// Block = 256 threads = 8 waves; wave owns a 32(M) x 64(N) strip.
// The 64x32 W tile is staged per block into a double-buffered LDS tile via
// CDNA5 async global->LDS copies (no VGPR staging, ASYNCcnt-tracked), so all
// 8 waves read it from LDS instead of fetching it redundantly from cache.
// MODE: 0 bias->bf16 | 1 bias->f32 | 2 plain->bf16 | 3 silu(aux)*acc->bf16
//       4 acc+auxF32->f32
// ---------------------------------------------------------------------------
template <int MODE>
__global__ __launch_bounds__(256) void gemm_bf16_kernel(const __bf16* __restrict__ A,
                                                        const __bf16* __restrict__ W,
                                                        const float* __restrict__ bias,
                                                        const void* __restrict__ aux,
                                                        float* __restrict__ outF,
                                                        __bf16* __restrict__ outB,
                                                        int M, int N, int K) {
  __shared__ __bf16 wt[2][64][WPAD];
  const int tid = threadIdx.x;
  const int lane = tid & 31;
  const int wave = tid >> 5;
  const int m0 = blockIdx.y * 256 + wave * 32;
  const int n0 = blockIdx.x * 64;
  const int r = lane & 15;
  const int hf = lane >> 4;
  const int wrow = tid >> 2;        // 0..63  (W tile row this thread stages)
  const int wseg = (tid & 3) * 8;   // 0,8,16,24 (8-element chunk)

  v8f acc[2][4];
#pragma unroll
  for (int mi = 0; mi < 2; ++mi)
#pragma unroll
    for (int t = 0; t < 4; ++t) acc[mi][t] = vzero8();

  const unsigned* arow0 = (const unsigned*)(A + (size_t)(m0 + r) * K);
  const unsigned* arow1 = (const unsigned*)(A + (size_t)(m0 + 16 + r) * K);
  const __bf16* wsrc = W + (size_t)(n0 + wrow) * K + wseg;
  const unsigned ldsOff0 = (unsigned)(uintptr_t)&wt[0][wrow][wseg];
  const unsigned ldsOff1 = (unsigned)(uintptr_t)&wt[1][wrow][wseg];

  // prologue: async-stage k0 = 0 into buffer 0
  async_ld16(ldsOff0, wsrc);
  wait_async0();
  __syncthreads();

  int cur = 0;
  for (int k0 = 0; k0 < K; k0 += 32) {
    const bool nxt = (k0 + 32) < K;
    if (nxt) async_ld16(cur ? ldsOff0 : ldsOff1, wsrc + k0 + 32);  // stage next tile
    if ((k0 + 64) < K) {
      __builtin_prefetch((const void*)(wsrc + k0 + 64), 0, 1);
      __builtin_prefetch((const void*)(arow0 + ((k0 + 64) >> 1)), 0, 1);
      __builtin_prefetch((const void*)(arow1 + ((k0 + 64) >> 1)), 0, 1);
    }
    union { v16bf v; unsigned u[8]; } a0, a1;
#pragma unroll
    for (int v = 0; v < 4; ++v) {
      const int p0 = (k0 + 2 * v + 16 * hf) >> 1;
      const int p1 = (k0 + 8 + 2 * v + 16 * hf) >> 1;
      a0.u[v] = arow0[p0];  a0.u[4 + v] = arow0[p1];
      a1.u[v] = arow1[p0];  a1.u[4 + v] = arow1[p1];
    }
#pragma unroll
    for (int t = 0; t < 4; ++t) {
      v16bf b = *(const v16bf*)&wt[cur][t * 16 + r][hf * 16];
      acc[0][t] = __builtin_amdgcn_wmma_f32_16x16x32_bf16(false, a0.v, false, b,
                                                          (short)0, acc[0][t], false, false);
      acc[1][t] = __builtin_amdgcn_wmma_f32_16x16x32_bf16(false, a1.v, false, b,
                                                          (short)0, acc[1][t], false, false);
    }
    if (nxt) wait_async0();   // own async copies have landed in LDS
    __syncthreads();          // everyone's copies landed; everyone done reading
    cur ^= 1;
  }

#pragma unroll
  for (int mi = 0; mi < 2; ++mi) {
#pragma unroll
    for (int t = 0; t < 4; ++t) {
      const int n = n0 + t * 16 + r;
      const float bv = (MODE == 0 || MODE == 1) ? bias[n] : 0.f;
#pragma unroll
      for (int j = 0; j < 8; ++j) {
        const int m = m0 + mi * 16 + 8 * hf + j;
        const size_t idx = (size_t)m * N + n;
        float v = acc[mi][t][j] + bv;
        if (MODE == 3) {
          const float g = bf2f(((const __bf16*)aux)[idx]);
          v = (g / (1.f + __expf(-g))) * v;   // silu(gate) * up
        }
        if (MODE == 4) v += ((const float*)aux)[idx];
        if (MODE == 1 || MODE == 4) outF[idx] = v;
        else                        outB[idx] = f2bf(v);
      }
    }
  }
}

// ---------------------------------------------------------------------------
// Coalesced LDS-tile transpose of the V section: vT[b][h][d][t].
// ---------------------------------------------------------------------------
__global__ __launch_bounds__(256) void vtrans_kernel(const __bf16* __restrict__ qkv,
                                                     __bf16* __restrict__ vT) {
  __shared__ __bf16 tile[64][72];
  const int blk = blockIdx.x;         // ((b*16+h)*32 + tchunk)
  const int tc = blk & 31;
  const int bh = blk >> 5;
  const int h = bh & 15;
  const int b = bh >> 4;
  const int t0 = tc * 64;
  const int c = threadIdx.x & 63;
  const int rr = threadIdx.x >> 6;    // 0..3
  const __bf16* src = qkv + 2 * DD + h * HD;
#pragma unroll
  for (int i = 0; i < 16; ++i) {
    const int tt = i * 4 + rr;
    tile[tt][c] = src[(size_t)(b * TT + t0 + tt) * (3 * DD) + c];  // coalesced along d
  }
  __syncthreads();
#pragma unroll
  for (int i = 0; i < 16; ++i) {
    const int d = i * 4 + rr;
    vT[((size_t)bh * HD + d) * TT + t0 + c] = tile[c][d];          // coalesced along t
  }
}

// ---------------------------------------------------------------------------
// Sliding-window attention: one wave per (b, h, 16-query block).
// Scores via WMMA -> f16 in LDS; softmax overwrites the same 16-bit slots
// with bf16 probabilities in place (one buffer, ~17.5 KB -> high occupancy);
// P@V via WMMA against vT. Output bf16 [B*T, D].
// ---------------------------------------------------------------------------
__global__ __launch_bounds__(32) void attn_kernel(const __bf16* __restrict__ qkv,
                                                  const __bf16* __restrict__ vT,
                                                  __bf16* __restrict__ attn_out) {
  __shared__ unsigned short sp[16][SCB];   // f16 scores, then bf16 probs (in place)
  __shared__ float rowsum[16];

  const int id = blockIdx.x;         // ((b*16+h)*128 + qb)
  const int qb = id & 127;
  const int bh = id >> 7;
  const int h = bh & 15;
  const int b = bh >> 4;
  const int lane = threadIdx.x;
  const int r = lane & 15;
  const int hf = lane >> 4;

  const int kb_start = (qb >= 32) ? (qb - 32) : 0;
  const int nkb = qb - kb_start + 1;   // <= 33
  const int nkeys = nkb * 16;
  const int nk32 = (nkeys + 31) & ~31;

  // Q A-fragments for both HD chunks (contiguous k = 16*hf + s per lane half).
  const unsigned* qrow =
      (const unsigned*)(qkv + (size_t)(b * TT + qb * 16 + r) * (3 * DD) + h * HD);
  union { v16bf v; unsigned u[8]; } a0, a1;
#pragma unroll
  for (int v = 0; v < 4; ++v) {
    const int p0 = (2 * v + 16 * hf) >> 1;
    const int p1 = (8 + 2 * v + 16 * hf) >> 1;
    a0.u[v] = qrow[p0];       a0.u[4 + v] = qrow[p1];
    a1.u[v] = qrow[16 + p0];  a1.u[4 + v] = qrow[16 + p1];
  }

  const int qi_base = qb * 16;
  for (int kb = 0; kb < nkb; ++kb) {
    const int key0 = (kb_start + kb) * 16;
    // B fragment: col = r (key), contraction = head dim, contiguous in memory.
    const __bf16* krow =
        qkv + (size_t)(b * TT + key0 + r) * (3 * DD) + DD + h * HD + 16 * hf;
    v16bf b0 = *(const v16bf*)krow;
    v16bf b1 = *(const v16bf*)(krow + 32);
    v8f acc = vzero8();
    acc = __builtin_amdgcn_wmma_f32_16x16x32_bf16(false, a0.v, false, b0, (short)0, acc, false, false);
    acc = __builtin_amdgcn_wmma_f32_16x16x32_bf16(false, a1.v, false, b1, (short)0, acc, false, false);
#pragma unroll
    for (int j = 0; j < 8; ++j) {
      const int m = j + 8 * hf;
      const int qi = qi_base + m;
      const int ki = key0 + r;
      const bool ok = (ki <= qi) && (qi - ki < WW);
      sp[m][kb * 16 + r] = f2h_bits(ok ? acc[j] * 0.125f : -6.0e4f);  // 1/sqrt(64)
    }
  }
  __syncthreads();

  // In-place softmax; lanes (r) and (r+16) split row r's columns; uniform
  // loop bounds keep all 32 lanes in lockstep (no exec masking).
  {
    float mx = -1e30f;
    for (int c0 = 0; c0 < nkeys; c0 += 2) mx = fmaxf(mx, h2f(sp[r][c0 + hf]));
    mx = fmaxf(mx, __shfl_xor(mx, 16));
    float s = 0.f;
    for (int c0 = 0; c0 < nkeys; c0 += 2) {
      const float e = __expf(h2f(sp[r][c0 + hf]) - mx);
      s += e;
      sp[r][c0 + hf] = bfbits(e);           // overwrite score with bf16 prob
    }
    s += __shfl_xor(s, 16);
    for (int c0 = nkeys; c0 < nk32; c0 += 2) sp[r][c0 + hf] = 0;  // zero pad
    if (hf == 0) rowsum[r] = s;
  }
  __syncthreads();

  // P @ V: A from LDS (A-frag layout), B from vT (contiguous along keys).
  v8f oacc[4];
#pragma unroll
  for (int t = 0; t < 4; ++t) oacc[t] = vzero8();

  for (int ks = 0; ks < nk32; ks += 32) {
    union { v16bf v; unsigned u[8]; } ap;
#pragma unroll
    for (int v = 0; v < 4; ++v) {
      const int k0 = ks + 2 * v + 16 * hf;
      const int k1 = ks + 8 + 2 * v + 16 * hf;
      ap.u[v]     = *(const unsigned*)&sp[r][k0];
      ap.u[4 + v] = *(const unsigned*)&sp[r][k1];
    }
    const int jglob = kb_start * 16 + ks;
#pragma unroll
    for (int t = 0; t < 4; ++t) {
      const int d = t * 16 + r;
      const __bf16* vrow = vT + ((size_t)bh * HD + d) * TT + jglob + 16 * hf;
      v16bf bv = *(const v16bf*)vrow;
      oacc[t] = __builtin_amdgcn_wmma_f32_16x16x32_bf16(false, ap.v, false, bv,
                                                        (short)0, oacc[t], false, false);
    }
  }
  __syncthreads();

#pragma unroll
  for (int t = 0; t < 4; ++t) {
    const int d = t * 16 + r;
#pragma unroll
    for (int j = 0; j < 8; ++j) {
      const int m = j + 8 * hf;
      const float o = oacc[t][j] / rowsum[m];
      attn_out[(size_t)(b * TT + qb * 16 + m) * DD + h * HD + d] = f2bf(o);
    }
  }
}

// ---------------------------------------------------------------------------
// Retrieval GEMV: ret_out[b][n] = rc[b] . ret_w[n]   (tiny: 4 MFLOP, fp32)
// ---------------------------------------------------------------------------
__global__ __launch_bounds__(256) void ret_kernel(const float* __restrict__ rc,
                                                  const float* __restrict__ retw,
                                                  float* __restrict__ ret_out) {
  const int n = blockIdx.x * 256 + threadIdx.x;   // 0..2047
  const int b = n >> 10, c = n & 1023;
  const float* rr = rc + (size_t)b * DD;
  const float* wr = retw + (size_t)c * DD;
  float s = 0.f;
  for (int k = 0; k < DD; ++k) s += rr[k] * wr[k];
  ret_out[n] = s;
}

// ---------------------------------------------------------------------------
// Gate softmax (channel 1 hard-masked) + mix + residual.
// ---------------------------------------------------------------------------
__global__ __launch_bounds__(256) void mix_kernel(const float* __restrict__ x,
                                                  const float* __restrict__ h,
                                                  const float* __restrict__ proj,
                                                  const float* __restrict__ ret,
                                                  const float* __restrict__ gatew,
                                                  const float* __restrict__ gateb,
                                                  float* __restrict__ x1) {
  __shared__ float red[8][2];
  const int row = blockIdx.x;
  const int b = row / TT;
  const float* hr = h + (size_t)row * DD;
  float s0 = 0.f, s2 = 0.f;
#pragma unroll
  for (int i = 0; i < 4; ++i) {
    const int c = threadIdx.x + i * 256;
    const float hv = hr[c];
    s0 += hv * gatew[c];
    s2 += hv * gatew[2 * DD + c];
  }
#pragma unroll
  for (int off = 16; off > 0; off >>= 1) {
    s0 += __shfl_xor(s0, off);
    s2 += __shfl_xor(s2, off);
  }
  const int lane = threadIdx.x & 31, wv = threadIdx.x >> 5;
  if (lane == 0) { red[wv][0] = s0; red[wv][1] = s2; }
  __syncthreads();
  if (threadIdx.x == 0) {
    float a = gateb[0], c2 = gateb[2];
    for (int i = 0; i < 8; ++i) { a += red[i][0]; c2 += red[i][1]; }
    const float mx = fmaxf(a, c2);
    const float e0 = __expf(a - mx), e2 = __expf(c2 - mx);
    const float inv = 1.f / (e0 + e2);     // channel 1 at -1e9 -> exp == 0
    red[0][0] = e0 * inv;
    red[0][1] = e2 * inv;
  }
  __syncthreads();
  const float g0 = red[0][0], g2 = red[0][1];
  const float* pr = proj + (size_t)row * DD;
  const float* rr = ret + (size_t)b * DD;
  const float* xr = x + (size_t)row * DD;
  float* o = x1 + (size_t)row * DD;
#pragma unroll
  for (int i = 0; i < 4; ++i) {
    const int c = threadIdx.x + i * 256;
    o[c] = xr[c] + g0 * pr[c] + g2 * rr[c];
  }
}

// ---------------------------------------------------------------------------
extern "C" void kernel_launch(void* const* d_in, const int* in_sizes, int n_in,
                              void* d_out, int out_size, void* d_ws, size_t ws_size,
                              hipStream_t stream) {
  const float* x    = (const float*)d_in[0];
  const float* rc   = (const float*)d_in[1];
  const float* n1w  = (const float*)d_in[2];
  const float* qkvw = (const float*)d_in[3];
  const float* qkvb = (const float*)d_in[4];
  const float* prjw = (const float*)d_in[5];
  const float* prjb = (const float*)d_in[6];
  const float* gtw  = (const float*)d_in[7];
  const float* gtb  = (const float*)d_in[8];
  const float* retw = (const float*)d_in[9];
  const float* n2w  = (const float*)d_in[10];
  const float* fgw  = (const float*)d_in[11];
  const float* fuw  = (const float*)d_in[12];
  const float* fdw  = (const float*)d_in[13];
  float* out = (float*)d_out;

  char* p = (char*)d_ws;
  auto alloc = [&](size_t bytes) -> char* {
    char* rp = p;
    p += (bytes + 255) & ~(size_t)255;
    return rp;
  };
  float*  h_f    = (float*) alloc((size_t)MTOT * DD * 4);
  __bf16* h_b    = (__bf16*)alloc((size_t)MTOT * DD * 2);
  __bf16* qkvw_b = (__bf16*)alloc((size_t)3 * DD * DD * 2);
  __bf16* prjw_b = (__bf16*)alloc((size_t)DD * DD * 2);
  __bf16* fgw_b  = (__bf16*)alloc((size_t)DFF_ * DD * 2);
  __bf16* fuw_b  = (__bf16*)alloc((size_t)DFF_ * DD * 2);
  __bf16* fdw_b  = (__bf16*)alloc((size_t)DD * DFF_ * 2);
  __bf16* qkv_b  = (__bf16*)alloc((size_t)MTOT * 3 * DD * 2);
  __bf16* vT_b   = (__bf16*)alloc((size_t)BB * NH * HD * TT * 2 + 256);  // +slack for padded P@V reads
  __bf16* attn_b = (__bf16*)alloc((size_t)MTOT * DD * 2);
  float*  proj_f = (float*) alloc((size_t)MTOT * DD * 4);
  float*  ret_f  = (float*) alloc((size_t)BB * DD * 4);
  float*  x1_f   = (float*) alloc((size_t)MTOT * DD * 4);
  __bf16* h2_b   = (__bf16*)alloc((size_t)MTOT * DD * 2);
  __bf16* G_b    = (__bf16*)alloc((size_t)MTOT * DFF_ * 2);
  __bf16* FF_b   = (__bf16*)alloc((size_t)MTOT * DFF_ * 2);

  auto cvt = [&](const float* src, __bf16* dst, int n) {
    f32_to_bf16_kernel<<<(n / 4 + 255) / 256, 256, 0, stream>>>(
        (const float4*)src, (uint2*)dst, n / 4);
  };
  cvt(qkvw, qkvw_b, 3 * DD * DD);
  cvt(prjw, prjw_b, DD * DD);
  cvt(fgw, fgw_b, DFF_ * DD);
  cvt(fuw, fuw_b, DFF_ * DD);
  cvt(fdw, fdw_b, DD * DFF_);

  // 1) pre-norm
  rmsnorm_kernel<true><<<MTOT, 256, 0, stream>>>(x, n1w, h_f, h_b);

  // 2) qkv = h @ qkv_w^T + qkv_b   (4096 x 3072 x 1024) -> bf16
  gemm_bf16_kernel<0><<<dim3(3 * DD / 64, MTOT / 256), 256, 0, stream>>>(
      h_b, qkvw_b, qkvb, nullptr, nullptr, qkv_b, MTOT, 3 * DD, DD);

  // 3) transpose V for attention B-fragments
  vtrans_kernel<<<BB * NH * (TT / 64), 256, 0, stream>>>(qkv_b, vT_b);

  // 4) sliding-window attention -> attn_b (bf16)
  attn_kernel<<<BB * NH * (TT / 16), 32, 0, stream>>>(qkv_b, vT_b, attn_b);

  // 5) output projection -> f32
  gemm_bf16_kernel<1><<<dim3(DD / 64, MTOT / 256), 256, 0, stream>>>(
      attn_b, prjw_b, prjb, nullptr, proj_f, nullptr, MTOT, DD, DD);

  // 6) retrieval stream + gated mix + residual
  ret_kernel<<<(BB * DD) / 256, 256, 0, stream>>>(rc, retw, ret_f);
  mix_kernel<<<MTOT, 256, 0, stream>>>(x, h_f, proj_f, ret_f, gtw, gtb, x1_f);

  // 7) second norm -> bf16 only
  rmsnorm_kernel<false><<<MTOT, 256, 0, stream>>>(x1_f, n2w, nullptr, h2_b);

  // 8) SwiGLU FFN
  gemm_bf16_kernel<2><<<dim3(DFF_ / 64, MTOT / 256), 256, 0, stream>>>(
      h2_b, fgw_b, nullptr, nullptr, nullptr, G_b, MTOT, DFF_, DD);
  gemm_bf16_kernel<3><<<dim3(DFF_ / 64, MTOT / 256), 256, 0, stream>>>(
      h2_b, fuw_b, nullptr, G_b, nullptr, FF_b, MTOT, DFF_, DD);
  gemm_bf16_kernel<4><<<dim3(DD / 64, MTOT / 256), 256, 0, stream>>>(
      FF_b, fdw_b, nullptr, x1_f, out, nullptr, MTOT, DD, DFF_);
}
